// Seq2seq_77610059038984
// MI455X (gfx1250) — matrix operand order repaired and checked by reading, not compiled, not run on hardware
//
#include <hip/hip_runtime.h>

// ---------------------------------------------------------------------------
// Seq2seq LSTM for MI455X (gfx1250): all GEMMs on v_wmma_f32_16x16x32_bf16.
// Weights pre-packed into WMMA B-fragment order (32x16 blocks, lane-contig).
// Time-parallel work (embeddings + input projections) hoisted out of the scan.
// ---------------------------------------------------------------------------

typedef __bf16 bf16;
typedef __attribute__((ext_vector_type(16))) __bf16 v16bf;
typedef __attribute__((ext_vector_type(8)))  __bf16 v8bf;
typedef __attribute__((ext_vector_type(8)))  float  v8f;

#define DEPTH   3
#define UNITS   128
#define EMBED   32
#define PATHS   2048
#define DEC_EMB 512
#define BATCH   256
#define TSTEPS  128
#define G4      512        // 4*UNITS

union AF { v16bf v; v8bf h[2]; };

__device__ __forceinline__ v8f vzero() {
    v8f z;
#pragma unroll
    for (int i = 0; i < 8; ++i) z[i] = 0.0f;
    return z;
}

__device__ __forceinline__ v8f wmma_bf16(v16bf a, v16bf b, v8f c) {
    return __builtin_amdgcn_wmma_f32_16x16x32_bf16(false, a, false, b,
                                                   (short)0, c, false, false);
}

__device__ __forceinline__ float sigm(float x) {
    return 1.0f / (1.0f + __expf(-x));
}

// --------------------------- weight pre-packing -----------------------------
// src: row-major [K][N] fp32.  dst: bf16 blocks of 32x16; block (kt,nt) holds
// lane L's 16 values: column n = nt*16 + L%16, K = kt*32+(L/16)*8+{0..7, 16..23}
__global__ __launch_bounds__(256) void pack_w(const float* __restrict__ src,
                                              bf16* __restrict__ dst,
                                              int K, int N) {
    int i = blockIdx.x * 256 + threadIdx.x;
    if (i >= K * N) return;
    int k = i / N, n = i % N;
    int kt = k >> 5, kin = k & 31, nt = n >> 4, nin = n & 15;
    int lane = ((kin >> 3) & 1) * 16 + nin;
    int pos  = (kin & 7) + ((kin >> 4) << 3);
    dst[(((size_t)kt * (N >> 4) + nt) * 32 + lane) * 16 + pos] = (bf16)src[i];
}

__global__ __launch_bounds__(256) void zero_f32(float* __restrict__ p, int n) {
    int i = blockIdx.x * 256 + threadIdx.x;
    if (i < n) p[i] = 0.0f;
}

// -------------------- embedding gather + input projection -------------------
// out[p][128] = relu(emb[idx[p]] @ W + b), K = EMBED = 32 (one WMMA k-step)
__global__ __launch_bounds__(256) void embed_proj_enc(
        const int* __restrict__ idx, const float* __restrict__ emb,
        const bf16* __restrict__ Wp, const float* __restrict__ bias,
        bf16* __restrict__ out) {
    const int tid = threadIdx.x, wave = tid >> 5, lane = tid & 31;
    const int half = lane >> 4, l16 = lane & 15;
    const int pb = (blockIdx.x * 8 + wave) * 16;       // 16 positions per wave
    const float* er = emb + (size_t)idx[pb + l16] * EMBED;
    const int kb = half * 8;
    AF a; v8bf lo, hi;
#pragma unroll
    for (int q = 0; q < 8; ++q) { lo[q] = (bf16)er[kb + q]; hi[q] = (bf16)er[kb + 16 + q]; }
    a.h[0] = lo; a.h[1] = hi;
    v8f acc[8];
#pragma unroll
    for (int nt = 0; nt < 8; ++nt) {
        v16bf b = *(const v16bf*)(Wp + ((size_t)nt * 32 + lane) * 16);
        acc[nt] = wmma_bf16(a.v, b, vzero());
    }
#pragma unroll
    for (int nt = 0; nt < 8; ++nt) {
        const int n = nt * 16 + l16;
#pragma unroll
        for (int j = 0; j < 8; ++j) {
            const int r = pb + half * 8 + j;
            out[(size_t)r * UNITS + n] = (bf16)fmaxf(acc[nt][j] + bias[n], 0.0f);
        }
    }
}

// same, K = DEC_EMB = 512 (16 WMMA k-steps)
__global__ __launch_bounds__(256) void embed_proj_dec(
        const int* __restrict__ idx, const float* __restrict__ emb,
        const bf16* __restrict__ Wp, const float* __restrict__ bias,
        bf16* __restrict__ out) {
    const int tid = threadIdx.x, wave = tid >> 5, lane = tid & 31;
    const int half = lane >> 4, l16 = lane & 15;
    const int pb = (blockIdx.x * 8 + wave) * 16;
    const float* er = emb + (size_t)idx[pb + l16] * DEC_EMB;
    v8f acc[8];
#pragma unroll
    for (int nt = 0; nt < 8; ++nt) acc[nt] = vzero();
    for (int kt = 0; kt < 16; ++kt) {
        const int kb = kt * 32 + half * 8;
        AF a; v8bf lo, hi;
#pragma unroll
        for (int q = 0; q < 8; ++q) { lo[q] = (bf16)er[kb + q]; hi[q] = (bf16)er[kb + 16 + q]; }
        a.h[0] = lo; a.h[1] = hi;
#pragma unroll
        for (int nt = 0; nt < 8; ++nt) {
            v16bf b = *(const v16bf*)(Wp + (((size_t)kt * 8 + nt) * 32 + lane) * 16);
            acc[nt] = wmma_bf16(a.v, b, acc[nt]);
        }
    }
#pragma unroll
    for (int nt = 0; nt < 8; ++nt) {
        const int n = nt * 16 + l16;
#pragma unroll
        for (int j = 0; j < 8; ++j) {
            const int r = pb + half * 8 + j;
            out[(size_t)r * UNITS + n] = (bf16)fmaxf(acc[nt][j] + bias[n], 0.0f);
        }
    }
}

// ---------------------- fused LSTM step (one layer) -------------------------
// z = x @ k + h @ rk + bias ;  i,f,g,o split ; c' = s(f)c + s(i)tanh(g) ;
// h' = s(o) tanh(c').  One WG = 16 batch rows x all 512 gate columns.
__global__ __launch_bounds__(256) void lstm_step(
        const bf16* __restrict__ A0, int lda0,         // layer input [B][*]
        const bf16* __restrict__ Wk,                   // packed [K=128][N=512]
        const bf16* __restrict__ Wrk,                  // packed [K=128][N=512]
        const float* __restrict__ bias,                // [512]
        float* __restrict__ c_state,                   // [B][128]
        bf16* __restrict__ h_state) {                  // [B][128] read+write
    __shared__ float zt[16 * G4];
    const int tid = threadIdx.x, wave = tid >> 5, lane = tid & 31;
    const int half = lane >> 4, l16 = lane & 15;
    const int mrow = blockIdx.x * 16;
    const int n0 = wave * 64;

    if (tid == 0) { __builtin_prefetch(Wk, 0, 1); __builtin_prefetch(Wrk, 0, 1); }

    v8f acc[4];
#pragma unroll
    for (int s = 0; s < 4; ++s) acc[s] = vzero();

    const bf16* Asrc[2] = { A0, h_state };
    const int   lda[2]  = { lda0, UNITS };
    const bf16* Wsrc[2] = { Wk, Wrk };

#pragma unroll
    for (int m = 0; m < 2; ++m) {
        const bf16* Arow = Asrc[m] + (size_t)(mrow + l16) * lda[m];
#pragma unroll
        for (int kt = 0; kt < 4; ++kt) {
            AF a;
            const int kb = kt * 32 + half * 8;
            a.h[0] = *(const v8bf*)(Arow + kb);
            a.h[1] = *(const v8bf*)(Arow + kb + 16);
#pragma unroll
            for (int s = 0; s < 4; ++s) {
                const int ntile = (n0 >> 4) + s;
                v16bf b = *(const v16bf*)(Wsrc[m] + (((size_t)kt * 32 + ntile) * 32 + lane) * 16);
                acc[s] = wmma_bf16(a.v, b, acc[s]);
            }
        }
    }
#pragma unroll
    for (int s = 0; s < 4; ++s) {
        const int n = n0 + s * 16 + l16;
#pragma unroll
        for (int j = 0; j < 8; ++j)
            zt[(half * 8 + j) * G4 + n] = acc[s][j];
    }
    __syncthreads();
    // gate nonlinearities + state update: 16 rows x 128 units
#pragma unroll
    for (int it = 0; it < 8; ++it) {
        const int e = tid + it * 256;
        const int r = e >> 7, u = e & 127;
        const float zi = zt[r * G4 + u]       + bias[u];
        const float zf = zt[r * G4 + 128 + u] + bias[128 + u];
        const float zg = zt[r * G4 + 256 + u] + bias[256 + u];
        const float zo = zt[r * G4 + 384 + u] + bias[384 + u];
        const int g = (mrow + r) * UNITS + u;
        const float cn = sigm(zf) * c_state[g] + sigm(zi) * tanhf(zg);
        c_state[g] = cn;
        h_state[g] = (bf16)(sigm(zo) * tanhf(cn));
    }
}

// -------------------- out1: relu(h @ W + b) -> bf16 [256x512] ---------------
__global__ __launch_bounds__(256) void gemm_bias_relu_bf16(
        const bf16* __restrict__ A, int lda,
        const bf16* __restrict__ Wp,          // packed K=128, N=512
        const float* __restrict__ bias,
        bf16* __restrict__ out) {
    const int tid = threadIdx.x, wave = tid >> 5, lane = tid & 31;
    const int half = lane >> 4, l16 = lane & 15;
    const int mrow = blockIdx.x * 16;
    const int n0 = wave * 64;
    v8f acc[4];
#pragma unroll
    for (int s = 0; s < 4; ++s) acc[s] = vzero();
    const bf16* Arow = A + (size_t)(mrow + l16) * lda;
#pragma unroll
    for (int kt = 0; kt < 4; ++kt) {
        AF a;
        const int kb = kt * 32 + half * 8;
        a.h[0] = *(const v8bf*)(Arow + kb);
        a.h[1] = *(const v8bf*)(Arow + kb + 16);
#pragma unroll
        for (int s = 0; s < 4; ++s) {
            const int ntile = (n0 >> 4) + s;
            v16bf b = *(const v16bf*)(Wp + (((size_t)kt * 32 + ntile) * 32 + lane) * 16);
            acc[s] = wmma_bf16(a.v, b, acc[s]);
        }
    }
#pragma unroll
    for (int s = 0; s < 4; ++s) {
        const int n = n0 + s * 16 + l16;
#pragma unroll
        for (int j = 0; j < 8; ++j) {
            const int row = mrow + half * 8 + j;
            out[(size_t)row * G4 + n] = (bf16)fmaxf(acc[s][j] + bias[n], 0.0f);
        }
    }
}

// ---------- out2: logits = out1 @ W + b  -> fp32, strided into d_out --------
__global__ __launch_bounds__(256) void gemm_bias_f32(
        const bf16* __restrict__ A,            // [256x512]
        const bf16* __restrict__ Wp,           // packed K=512, N=2048
        const float* __restrict__ bias,        // [2048]
        float* __restrict__ out, int ldo) {
    const int tid = threadIdx.x, wave = tid >> 5, lane = tid & 31;
    const int half = lane >> 4, l16 = lane & 15;
    const int mb = blockIdx.x >> 3;            // 16 m-blocks
    const int nseg = blockIdx.x & 7;           // 8 n-segments of 256
    const int n0 = nseg * 256 + wave * 32;
    v8f acc[2];
    acc[0] = vzero(); acc[1] = vzero();
    const bf16* Arow = A + (size_t)(mb * 16 + l16) * DEC_EMB;
#pragma unroll
    for (int kt = 0; kt < 16; ++kt) {
        AF a;
        const int kb = kt * 32 + half * 8;
        a.h[0] = *(const v8bf*)(Arow + kb);
        a.h[1] = *(const v8bf*)(Arow + kb + 16);
#pragma unroll
        for (int s = 0; s < 2; ++s) {
            const int ntile = (n0 >> 4) + s;
            v16bf b = *(const v16bf*)(Wp + (((size_t)kt * 128 + ntile) * 32 + lane) * 16);
            acc[s] = wmma_bf16(a.v, b, acc[s]);
        }
    }
#pragma unroll
    for (int s = 0; s < 2; ++s) {
        const int n = n0 + s * 16 + l16;
#pragma unroll
        for (int j = 0; j < 8; ++j) {
            const int row = mb * 16 + half * 8 + j;
            out[(size_t)row * ldo + n] = acc[s][j] + bias[n];
        }
    }
}

// ------------------------- row softmax over 2048 ----------------------------
__global__ __launch_bounds__(256) void softmax_rows(float* __restrict__ p, int ldo) {
    float* row = p + (size_t)blockIdx.x * ldo;
    __shared__ float red[256];
    const int tid = threadIdx.x;
    float m = -1e30f;
    for (int i = tid; i < PATHS; i += 256) m = fmaxf(m, row[i]);
    red[tid] = m; __syncthreads();
    for (int s = 128; s > 0; s >>= 1) {
        if (tid < s) red[tid] = fmaxf(red[tid], red[tid + s]);
        __syncthreads();
    }
    m = red[0]; __syncthreads();
    float sum = 0.0f;
    for (int i = tid; i < PATHS; i += 256) {
        float e = __expf(row[i] - m);
        row[i] = e; sum += e;
    }
    red[tid] = sum; __syncthreads();
    for (int s = 128; s > 0; s >>= 1) {
        if (tid < s) red[tid] += red[tid + s];
        __syncthreads();
    }
    const float inv = 1.0f / red[0];
    for (int i = tid; i < PATHS; i += 256) row[i] *= inv;
}

// ---------------------------------------------------------------------------
extern "C" void kernel_launch(void* const* d_in, const int* in_sizes, int n_in,
                              void* d_out, int out_size, void* d_ws, size_t ws_size,
                              hipStream_t stream) {
    const int*   inputs   = (const int*)  d_in[0];
    const int*   targets  = (const int*)  d_in[1];
    const float* emb_in   = (const float*)d_in[2];
    const float* enc_W    = (const float*)d_in[3];
    const float* enc_b    = (const float*)d_in[4];
    const float* enc_k    = (const float*)d_in[5];
    const float* enc_rk   = (const float*)d_in[6];
    const float* enc_bias = (const float*)d_in[7];
    const float* emb_dec  = (const float*)d_in[8];
    const float* dec_W    = (const float*)d_in[9];
    const float* dec_b    = (const float*)d_in[10];
    const float* dec_k    = (const float*)d_in[11];
    const float* dec_rk   = (const float*)d_in[12];
    const float* dec_bias = (const float*)d_in[13];
    const float* out1_W   = (const float*)d_in[14];
    const float* out1_b   = (const float*)d_in[15];
    const float* out2_W   = (const float*)d_in[16];
    const float* out2_b   = (const float*)d_in[17];
    float* out = (float*)d_out;

    char* ws = (char*)d_ws;
    size_t off = 0;
    auto take = [&](size_t bytes) -> char* {
        char* p = ws + off;
        off = (off + bytes + 255) & ~(size_t)255;
        return p;
    };
    const size_t LW = 128 * 512;                 // per-layer lstm weight elems
    bf16* enc_kp  = (bf16*)take(3 * LW * 2);
    bf16* enc_rkp = (bf16*)take(3 * LW * 2);
    bf16* dec_kp  = (bf16*)take(3 * LW * 2);
    bf16* dec_rkp = (bf16*)take(3 * LW * 2);
    bf16* encWp   = (bf16*)take(32 * 128 * 2);
    bf16* decWp   = (bf16*)take(512 * 128 * 2);
    bf16* out1Wp  = (bf16*)take(128 * 512 * 2);
    bf16* out2Wp  = (bf16*)take((size_t)512 * 2048 * 2);
    bf16* xe      = (bf16*)take((size_t)BATCH * TSTEPS * UNITS * 2);
    bf16* xd      = (bf16*)take((size_t)BATCH * TSTEPS * UNITS * 2);
    float* cst    = (float*)take((size_t)3 * BATCH * UNITS * 4);
    bf16* hst     = (bf16*)take((size_t)3 * BATCH * UNITS * 2);
    bf16* out1buf = (bf16*)take((size_t)BATCH * 512 * 2);

    const size_t LSTATE = (size_t)BATCH * UNITS; // 32768

    // ---- pack weights to WMMA fragment order (bf16) ----
    for (int l = 0; l < 3; ++l) {
        pack_w<<<256, 256, 0, stream>>>(enc_k  + l * LW, enc_kp  + l * LW, 128, 512);
        pack_w<<<256, 256, 0, stream>>>(enc_rk + l * LW, enc_rkp + l * LW, 128, 512);
        pack_w<<<256, 256, 0, stream>>>(dec_k  + l * LW, dec_kp  + l * LW, 128, 512);
        pack_w<<<256, 256, 0, stream>>>(dec_rk + l * LW, dec_rkp + l * LW, 128, 512);
    }
    pack_w<<<16,   256, 0, stream>>>(enc_W,  encWp,  32,  128);
    pack_w<<<256,  256, 0, stream>>>(dec_W,  decWp,  512, 128);
    pack_w<<<256,  256, 0, stream>>>(out1_W, out1Wp, 128, 512);
    pack_w<<<4096, 256, 0, stream>>>(out2_W, out2Wp, 512, 2048);

    // ---- zero initial state (every call -> deterministic) ----
    zero_f32<<<384, 256, 0, stream>>>(cst, 3 * (int)LSTATE);
    zero_f32<<<192, 256, 0, stream>>>((float*)hst, 3 * (int)LSTATE / 2);

    // ---- time-parallel: embeddings + input projections for all steps ----
    embed_proj_enc<<<256, 256, 0, stream>>>(inputs,  emb_in,  encWp, enc_b, xe);
    embed_proj_dec<<<256, 256, 0, stream>>>(targets, emb_dec, decWp, dec_b, xd);

    // ---- encoder scan ----
    for (int t = 0; t < TSTEPS; ++t) {
        lstm_step<<<16, 256, 0, stream>>>(xe + (size_t)t * UNITS, TSTEPS * UNITS,
                                          enc_kp, enc_rkp, enc_bias,
                                          cst, hst);
        lstm_step<<<16, 256, 0, stream>>>(hst, UNITS,
                                          enc_kp + LW, enc_rkp + LW, enc_bias + 512,
                                          cst + LSTATE, hst + LSTATE);
        lstm_step<<<16, 256, 0, stream>>>(hst + LSTATE, UNITS,
                                          enc_kp + 2 * LW, enc_rkp + 2 * LW, enc_bias + 1024,
                                          cst + 2 * LSTATE, hst + 2 * LSTATE);
    }

    // ---- decoder scan (state carries over from encoder) ----
    for (int t = 0; t < TSTEPS; ++t) {
        lstm_step<<<16, 256, 0, stream>>>(xd + (size_t)t * UNITS, TSTEPS * UNITS,
                                          dec_kp, dec_rkp, dec_bias,
                                          cst, hst);
        lstm_step<<<16, 256, 0, stream>>>(hst, UNITS,
                                          dec_kp + LW, dec_rkp + LW, dec_bias + 512,
                                          cst + LSTATE, hst + LSTATE);
        lstm_step<<<16, 256, 0, stream>>>(hst + LSTATE, UNITS,
                                          dec_kp + 2 * LW, dec_rkp + 2 * LW, dec_bias + 1024,
                                          cst + 2 * LSTATE, hst + 2 * LSTATE);
        gemm_bias_relu_bf16<<<16, 256, 0, stream>>>(hst + 2 * LSTATE, UNITS,
                                                    out1Wp, out1_b, out1buf);
        gemm_bias_f32<<<128, 256, 0, stream>>>(out1buf, out2Wp, out2_b,
                                               out + (size_t)t * PATHS, TSTEPS * PATHS);
        softmax_rows<<<256, 256, 0, stream>>>(out + (size_t)t * PATHS, TSTEPS * PATHS);
    }
}
